// NeRFRenderer_56530359550738
// MI455X (gfx1250) — matrix operand order repaired and checked by reading, not compiled, not run on hardware
//
#include <hip/hip_runtime.h>
#include <hip/hip_bf16.h>

typedef __attribute__((ext_vector_type(16))) _Float16 v16h;
typedef __attribute__((ext_vector_type(8)))  _Float16 v8h;
typedef __attribute__((ext_vector_type(8)))  float    v8f;

#define R_RAYS   4096
#define NSTEP    64
#define SFINE    128
#define EPS_FD   0.005f

__device__ __forceinline__ float sigmoidf_(float x) { return 1.0f / (1.0f + __expf(-x)); }
__device__ __forceinline__ float clampf_(float x, float lo, float hi) { return fminf(fmaxf(x, lo), hi); }

__device__ __forceinline__ v16h combine16(v8h a0, v8h a1) {
  v16h a;
#pragma unroll
  for (int i = 0; i < 8; ++i) { a[i] = a0[i]; a[i + 8] = a1[i]; }
  return a;
}

// A-matrix 16x32 f16 fragment (CDNA5 ISA 7.12.2): lane's 16 elements are two
// contiguous 8-half runs of row m = lane&15: halfs [hi*8, hi*8+8) and [16+hi*8, ...).
// -> two ds_load_b128 per lane (lda in halfs; 16B alignment guaranteed by callers).
__device__ __forceinline__ v16h ld_a(const _Float16* A, int lda) {
  int lane = threadIdx.x & 31;
  int m = lane & 15, hi = lane >> 4;
  const v8h* p = (const v8h*)(A + m * lda + hi * 8);
  const v8h* q = (const v8h*)(A + m * lda + 16 + hi * 8);
  return combine16(*p, *q);
}

// B fragments are staged pre-swizzled: 32 lanes x 16 contiguous halfs.
__device__ __forceinline__ v16h ld_bf(const _Float16* F) {
  int lane = threadIdx.x & 31;
  const v8h* p = (const v8h*)(F + lane * 16);
  return combine16(p[0], p[1]);
}

// Stage one 32x16 B tile of a row-major float matrix W (rows x cols, leading dim ldw)
// into WMMA B-fragment order with zero padding. Frag element e of lane maps to
// k = (lane>>4)*8 + (e&7) + (e>>3)*16, n = lane&15.
__device__ __forceinline__ void stage_b(_Float16* F, const float* W, int ldw,
                                        int rows, int cols, int row0, int col0,
                                        int tid, int nthreads) {
  for (int i = tid; i < 512; i += nthreads) {
    int lane = i >> 4, e = i & 15;
    int hi = lane >> 4, n = lane & 15;
    int k = (hi << 3) + (e & 7) + ((e >> 3) << 4);
    int gr = row0 + k, gc = col0 + n;
    float v = (gr < rows && gc < cols) ? W[gr * ldw + gc] : 0.0f;
    F[i] = (_Float16)v;
  }
}

__device__ __forceinline__ v8f wmma32(v16h a, v16h b, v8f c) {
  return __builtin_amdgcn_wmma_f32_16x16x32_f16(false, a, false, b, (short)0, c, false, false);
}

// ---------------------------------------------------------------------------
// Kernel 1: per-ray AABB near/far and coarse z linspace
// ---------------------------------------------------------------------------
__global__ void prep_kernel(const float* __restrict__ ro, const float* __restrict__ rd,
                            float* __restrict__ zc, float* __restrict__ nearb,
                            float* __restrict__ farb, float* __restrict__ sdistb) {
  int r = blockIdx.x * blockDim.x + threadIdx.x;
  if (r >= R_RAYS) return;
  float nearv = -1e30f, farv = 1e30f;
#pragma unroll
  for (int a = 0; a < 3; ++a) {
    float o = ro[r * 3 + a];
    float d = rd[r * 3 + a] + 1e-15f;
    float t0 = (-1.0f - o) / d;
    float t1 = ( 1.0f - o) / d;
    nearv = fmaxf(nearv, fminf(t0, t1));
    farv  = fminf(farv,  fmaxf(t0, t1));
  }
  nearv = fmaxf(nearv, 0.05f);
  float sd = (farv - nearv) / (float)NSTEP;
  nearb[r] = nearv; farb[r] = farv; sdistb[r] = sd;
  for (int i = 0; i < NSTEP; ++i)
    zc[r * NSTEP + i] = nearv + (farv - nearv) * ((float)i * (1.0f / 63.0f));
}

// ---------------------------------------------------------------------------
// Kernel 2: coarse SDF (layer1 via WMMA, layer2 col0 via VALU dot)
// one wave per tile of 16 points; 8 waves per block
// ---------------------------------------------------------------------------
__global__ __launch_bounds__(256) void coarse_kernel(
    const float* __restrict__ ro, const float* __restrict__ rd,
    const float* __restrict__ W1, const float* __restrict__ b1,
    const float* __restrict__ W2, const float* __restrict__ b2,
    const float* __restrict__ zc, float* __restrict__ sdf_out) {
  __shared__ alignas(16) _Float16 sW1f[4][512];
  __shared__ float sw2c0[64];
  __shared__ float sb1[64];
  __shared__ float sb20;
  __shared__ alignas(16) _Float16 sA[8][16 * 32];
  __shared__ alignas(16) _Float16 sH[8][16 * 64];

  int tid = threadIdx.x;
#pragma unroll
  for (int nt = 0; nt < 4; ++nt)
    stage_b(sW1f[nt], W1, 64, 3, 64, 0, nt * 16, tid, 256);
  for (int i = tid; i < 64; i += blockDim.x) { sw2c0[i] = W2[i * 65]; sb1[i] = b1[i]; }
  if (tid == 0) sb20 = b2[0];
  __syncthreads();

  int w = tid >> 5, lane = tid & 31;
  int tile = blockIdx.x * 8 + w;
  int p0 = tile * 16;
  _Float16* A = sA[w];
  _Float16* H = sH[w];

  for (int i = lane; i < 16 * 32; i += 32) A[i] = (_Float16)0.0f;

  if (lane < 16) {
    int p = p0 + lane;
    int r = p >> 6;
    float z = zc[p];
    A[lane * 32 + 0] = (_Float16)clampf_(ro[r * 3 + 0] + rd[r * 3 + 0] * z, -1.0f, 1.0f);
    A[lane * 32 + 1] = (_Float16)clampf_(ro[r * 3 + 1] + rd[r * 3 + 1] * z, -1.0f, 1.0f);
    A[lane * 32 + 2] = (_Float16)clampf_(ro[r * 3 + 2] + rd[r * 3 + 2] * z, -1.0f, 1.0f);
  }

  int n = lane & 15, hi = lane >> 4;
  v16h a = ld_a(A, 32);
#pragma unroll
  for (int nt = 0; nt < 4; ++nt) {
    v16h b = ld_bf(sW1f[nt]);
    v8f acc = {};
    acc = wmma32(a, b, acc);
    float bb = sb1[nt * 16 + n];
#pragma unroll
    for (int i2 = 0; i2 < 8; ++i2) {
      float h = fmaxf(acc[i2] + bb, 0.0f);
      H[(i2 + 8 * hi) * 64 + nt * 16 + n] = (_Float16)h;
    }
  }

  if (lane < 16) {
    float s = sb20;
#pragma unroll 8
    for (int k = 0; k < 64; ++k) s += (float)H[lane * 64 + k] * sw2c0[k];
    sdf_out[p0 + lane] = s;
  }
}

// ---------------------------------------------------------------------------
// Kernel 3: per-ray importance sampling + merge of coarse/fine z values
// ---------------------------------------------------------------------------
__global__ void pdf_kernel(const float* __restrict__ zc, const float* __restrict__ sdfc,
                           const float* __restrict__ sdistb,
                           float* __restrict__ deltas, float* __restrict__ zmid) {
  int r = blockIdx.x * blockDim.x + threadIdx.x;
  if (r >= R_RAYS) return;

  float z[NSTEP], sl[NSTEP];
  for (int i = 0; i < NSTEP; ++i) { z[i] = zc[r * NSTEP + i]; sl[i] = sdfc[r * NSTEP + i]; }

  float wc[NSTEP - 1];
  float T = 1.0f, wsum = 0.0f;
  for (int j = 0; j < NSTEP - 1; ++j) {
    float d = z[j + 1] - z[j];
    float mid = 0.5f * (sl[j + 1] + sl[j]);
    float cosv = clampf_((sl[j + 1] - sl[j]) / (d + 1e-5f), -1000.0f, 0.0f);
    float pc = sigmoidf_((mid - cosv * d * 0.5f) * 64.0f);
    float nc = sigmoidf_((mid + cosv * d * 0.5f) * 64.0f);
    float a = clampf_((pc - nc + 1e-5f) / (pc + 1e-5f), 0.0f, 1.0f);
    float wv = a * T;
    T *= (1.0f - a + 1e-7f);
    wc[j] = wv + 1e-5f;
    wsum += wc[j];
  }
  float cdf[NSTEP];
  cdf[0] = 0.0f;
  float acc = 0.0f;
  for (int j = 0; j < NSTEP - 1; ++j) { acc += wc[j] / wsum; cdf[j + 1] = acc; }

  float nz[NSTEP];
  int ind = 0;
  for (int j = 0; j < NSTEP; ++j) {
    float u = (1.0f / 128.0f) + (float)j * (1.0f / 64.0f);
    while (ind < NSTEP && cdf[ind] <= u) ++ind;
    int below = min(max(ind - 1, 0), NSTEP - 1);
    int above = min(ind, NSTEP - 1);
    float cb = cdf[below], ca = cdf[above];
    float den = ca - cb;
    if (den < 1e-5f) den = 1.0f;
    float t = (u - cb) / den;
    nz[j] = z[below] + t * (z[above] - z[below]);
  }

  float za[SFINE];
  int i = 0, j2 = 0;
  for (int k = 0; k < SFINE; ++k) {
    float av = (i < NSTEP) ? z[i] : 3.4e38f;
    float bv = (j2 < NSTEP) ? nz[j2] : 3.4e38f;
    if (av <= bv) { za[k] = av; ++i; } else { za[k] = bv; ++j2; }
  }
  float sd = sdistb[r];
  for (int k = 0; k < SFINE - 1; ++k) {
    float dl = za[k + 1] - za[k];
    deltas[r * SFINE + k] = dl;
    zmid[r * SFINE + k] = za[k] + 0.5f * dl;
  }
  deltas[r * SFINE + SFINE - 1] = sd;
  zmid[r * SFINE + SFINE - 1] = za[SFINE - 1];
}

// ---------------------------------------------------------------------------
// Kernel 4: fused fine pass — 7x SDF net (center + FD grads), color net, alpha
// one wave per 16-sample tile; 4 waves per block
// ---------------------------------------------------------------------------
__global__ __launch_bounds__(128) void fine_kernel(
    const float* __restrict__ ro, const float* __restrict__ rd,
    const float* __restrict__ W1, const float* __restrict__ b1,
    const float* __restrict__ W2, const float* __restrict__ b2,
    const float* __restrict__ Wc1, const float* __restrict__ bc1,
    const float* __restrict__ Wc2, const float* __restrict__ bc2,
    const float* __restrict__ inv_s_param,
    const float* __restrict__ zmid, const float* __restrict__ deltas,
    float* __restrict__ alpha_out, float* __restrict__ color_out) {
  // B-fragment-order weight tiles (pre-swizzled)
  __shared__ alignas(16) _Float16 sW1f[4][512];        // W1 3x64 -> 4 N-tiles
  __shared__ alignas(16) _Float16 sW2f[5][2][512];     // W2 64x65 -> 5 N x 2 K
  __shared__ alignas(16) _Float16 sWc1f[4][3][512];    // Wc1 73x64 -> 4 N x 3 K
  __shared__ alignas(16) _Float16 sWc2f[2][512];       // Wc2 64x3 -> 1 N x 2 K
  __shared__ float sb1[64], sb2[80], sbc1[64], sbc2[16];
  __shared__ alignas(16) _Float16 sA[4][16 * 96];      // input / cin tile per wave
  __shared__ alignas(16) _Float16 sH[4][16 * 64];      // hidden per wave
  __shared__ alignas(16) _Float16 sF[4][16 * 64];      // feat per wave
  __shared__ float sSdf[4][16], sTmp[4][16];
  __shared__ float sGrad[4][48], sPts[4][48], sDir[4][48], sDel[4][16];

  int tid = threadIdx.x;
#pragma unroll
  for (int nt = 0; nt < 4; ++nt)
    stage_b(sW1f[nt], W1, 64, 3, 64, 0, nt * 16, tid, 128);
#pragma unroll
  for (int nt = 0; nt < 5; ++nt)
#pragma unroll
    for (int kc = 0; kc < 2; ++kc)
      stage_b(sW2f[nt][kc], W2, 65, 64, 65, kc * 32, nt * 16, tid, 128);
#pragma unroll
  for (int nt = 0; nt < 4; ++nt)
#pragma unroll
    for (int kc = 0; kc < 3; ++kc)
      stage_b(sWc1f[nt][kc], Wc1, 64, 73, 64, kc * 32, nt * 16, tid, 128);
#pragma unroll
  for (int kc = 0; kc < 2; ++kc)
    stage_b(sWc2f[kc], Wc2, 3, 64, 3, kc * 32, 0, tid, 128);
  for (int i = tid; i < 64; i += blockDim.x) { sb1[i] = b1[i]; sbc1[i] = bc1[i]; }
  for (int i = tid; i < 80; i += blockDim.x) sb2[i] = (i < 65) ? b2[i] : 0.0f;
  for (int i = tid; i < 16; i += blockDim.x) sbc2[i] = (i < 3) ? bc2[i] : 0.0f;
  __syncthreads();

  int w = tid >> 5, lane = tid & 31;
  int n = lane & 15, hi = lane >> 4;
  int tile = blockIdx.x * 4 + w;
  int s0 = tile * 16;
  _Float16* A  = sA[w];
  _Float16* H  = sH[w];
  _Float16* Ft = sF[w];

  for (int i = lane; i < 16 * 96; i += 32) A[i] = (_Float16)0.0f;

  if (lane < 16) {
    int s = s0 + lane;
    int r = s >> 7;
    float zm = zmid[s];
    float dx = rd[r * 3 + 0], dy = rd[r * 3 + 1], dz = rd[r * 3 + 2];
    sPts[w][lane * 3 + 0] = clampf_(ro[r * 3 + 0] + dx * zm, -1.0f, 1.0f);
    sPts[w][lane * 3 + 1] = clampf_(ro[r * 3 + 1] + dy * zm, -1.0f, 1.0f);
    sPts[w][lane * 3 + 2] = clampf_(ro[r * 3 + 2] + dz * zm, -1.0f, 1.0f);
    sDir[w][lane * 3 + 0] = dx; sDir[w][lane * 3 + 1] = dy; sDir[w][lane * 3 + 2] = dz;
    sDel[w][lane] = deltas[s];
  }

  // Hoist loop-invariant B fragments into registers
  v16h bW1[4];
#pragma unroll
  for (int nt = 0; nt < 4; ++nt) bW1[nt] = ld_bf(sW1f[nt]);
  v16h bW2c0[2];
#pragma unroll
  for (int kc = 0; kc < 2; ++kc) bW2c0[kc] = ld_bf(sW2f[0][kc]);

  // ---- 7 SDF evaluations: e=0 center; e=1..6 FD offsets (+x,-x,+y,-y,+z,-z)
  for (int e = 0; e < 7; ++e) {
    int axis = (e > 0) ? ((e - 1) >> 1) : 0;
    float sgn = (e == 0) ? 0.0f : ((e & 1) ? 1.0f : -1.0f);
    if (lane < 16) {
      float p0v = sPts[w][lane * 3 + 0];
      float p1v = sPts[w][lane * 3 + 1];
      float p2v = sPts[w][lane * 3 + 2];
      if (e > 0) {
        if (axis == 0) p0v += sgn * EPS_FD;
        else if (axis == 1) p1v += sgn * EPS_FD;
        else p2v += sgn * EPS_FD;
      }
      A[lane * 96 + 0] = (_Float16)p0v;
      A[lane * 96 + 1] = (_Float16)p1v;
      A[lane * 96 + 2] = (_Float16)p2v;
    }

    // layer1: (16x32)x(32x64), shared A fragment across 4 N-tiles
    {
      v16h a = ld_a(A, 96);
#pragma unroll
      for (int nt = 0; nt < 4; ++nt) {
        v8f acc = {};
        acc = wmma32(a, bW1[nt], acc);
        float bb = sb1[nt * 16 + n];
#pragma unroll
        for (int i2 = 0; i2 < 8; ++i2) {
          float h = fmaxf(acc[i2] + bb, 0.0f);
          H[(i2 + 8 * hi) * 64 + nt * 16 + n] = (_Float16)h;
        }
      }
    }

    // layer2: (16x64)x(64x65pad80); A fragments of H shared across N-tiles
    v16h aH0 = ld_a(&H[0], 64);
    v16h aH1 = ld_a(&H[32], 64);
    int ntmax = (e == 0) ? 5 : 1;
    for (int nt = 0; nt < ntmax; ++nt) {
      v16h b0 = (nt == 0) ? bW2c0[0] : ld_bf(sW2f[nt][0]);
      v16h b1 = (nt == 0) ? bW2c0[1] : ld_bf(sW2f[nt][1]);
      v8f acc = {};
      acc = wmma32(aH0, b0, acc);
      acc = wmma32(aH1, b1, acc);
#pragma unroll
      for (int i2 = 0; i2 < 8; ++i2) {
        int m = i2 + 8 * hi;
        int c = nt * 16 + n;
        float val = acc[i2] + sb2[c];
        if (e == 0) {
          if (c == 0) sSdf[w][m] = val;
          else if (c <= 64) Ft[m * 64 + (c - 1)] = (_Float16)val;
        } else {
          if (c == 0) {
            if (e & 1) sTmp[w][m] = val;                            // plus pass
            else sGrad[w][m * 3 + axis] =
                     (sTmp[w][m] - val) * (1.0f / (2.0f * EPS_FD)); // minus pass
          }
        }
      }
    }
  }

  // ---- build cin = [pts, dirs, normal, feat] (73 cols, padded to 96)
  if (lane < 16) {
    float gx = sGrad[w][lane * 3 + 0];
    float gy = sGrad[w][lane * 3 + 1];
    float gz = sGrad[w][lane * 3 + 2];
    float inv = 1.0f / (1e-5f + sqrtf(gx * gx + gy * gy + gz * gz));
    float nx = gx * inv, ny = gy * inv, nz = gz * inv;
    A[lane * 96 + 0] = (_Float16)sPts[w][lane * 3 + 0];
    A[lane * 96 + 1] = (_Float16)sPts[w][lane * 3 + 1];
    A[lane * 96 + 2] = (_Float16)sPts[w][lane * 3 + 2];
    A[lane * 96 + 3] = (_Float16)sDir[w][lane * 3 + 0];
    A[lane * 96 + 4] = (_Float16)sDir[w][lane * 3 + 1];
    A[lane * 96 + 5] = (_Float16)sDir[w][lane * 3 + 2];
    A[lane * 96 + 6] = (_Float16)nx;
    A[lane * 96 + 7] = (_Float16)ny;
    A[lane * 96 + 8] = (_Float16)nz;
    for (int k = 0; k < 64; ++k) A[lane * 96 + 9 + k] = Ft[lane * 64 + k];
    sGrad[w][lane * 3 + 0] = nx;  // keep normal for alpha stage
    sGrad[w][lane * 3 + 1] = ny;
    sGrad[w][lane * 3 + 2] = nz;
  }

  // ---- color layer1: (16x96)x(96x64), A fragments shared across N-tiles
  {
    v16h aC0 = ld_a(&A[0], 96);
    v16h aC1 = ld_a(&A[32], 96);
    v16h aC2 = ld_a(&A[64], 96);
#pragma unroll
    for (int nt = 0; nt < 4; ++nt) {
      v8f acc = {};
      acc = wmma32(aC0, ld_bf(sWc1f[nt][0]), acc);
      acc = wmma32(aC1, ld_bf(sWc1f[nt][1]), acc);
      acc = wmma32(aC2, ld_bf(sWc1f[nt][2]), acc);
      float bb = sbc1[nt * 16 + n];
#pragma unroll
      for (int i2 = 0; i2 < 8; ++i2) {
        float h = fmaxf(acc[i2] + bb, 0.0f);
        H[(i2 + 8 * hi) * 64 + nt * 16 + n] = (_Float16)h;
      }
    }
  }

  // ---- color layer2: (16x64)x(64x3pad16)
  {
    v16h aH0 = ld_a(&H[0], 64);
    v16h aH1 = ld_a(&H[32], 64);
    v8f acc = {};
    acc = wmma32(aH0, ld_bf(sWc2f[0]), acc);
    acc = wmma32(aH1, ld_bf(sWc2f[1]), acc);
    if (n < 3) {
#pragma unroll
      for (int i2 = 0; i2 < 8; ++i2) {
        int m = i2 + 8 * hi;
        color_out[(s0 + m) * 3 + n] = sigmoidf_(acc[i2] + sbc2[n]);
      }
    }
  }

  // ---- alpha (NeuS transmittance terms)
  if (lane < 16) {
    float inv_s = __expf(10.0f * inv_s_param[0]);
    float sdf = sSdf[w][lane];
    float nx = sGrad[w][lane * 3 + 0], ny = sGrad[w][lane * 3 + 1], nz = sGrad[w][lane * 3 + 2];
    float dx = sDir[w][lane * 3 + 0], dy = sDir[w][lane * 3 + 1], dz = sDir[w][lane * 3 + 2];
    float tc = dx * nx + dy * ny + dz * nz;
    float x = -100.0f * tc;                       // softplus(100*(-tc))/100
    float sp = (fmaxf(x, 0.0f) + log1pf(__expf(-fabsf(x)))) * 0.01f;
    float iter_cos = -sp;
    float dfl = sDel[w][lane];
    float pc = sigmoidf_((sdf - iter_cos * dfl * 0.5f) * inv_s);
    float nc = sigmoidf_((sdf + iter_cos * dfl * 0.5f) * inv_s);
    alpha_out[s0 + lane] = clampf_((pc - nc + 1e-5f) / (pc + 1e-5f), 0.0f, 1.0f);
  }
}

// ---------------------------------------------------------------------------
// Kernel 5: per-ray compositing (exclusive cumprod) + white background
// ---------------------------------------------------------------------------
__global__ void composite_kernel(const float* __restrict__ alpha,
                                 const float* __restrict__ colr,
                                 float* __restrict__ out) {
  int r = blockIdx.x * blockDim.x + threadIdx.x;
  if (r >= R_RAYS) return;
  float T = 1.0f, wsum = 0.0f, im0 = 0.0f, im1 = 0.0f, im2 = 0.0f;
  for (int s = 0; s < SFINE; ++s) {
    float a = alpha[r * SFINE + s];
    float wv = a * T;
    T *= (1.0f - a + 1e-7f);
    const float* c = &colr[(r * SFINE + s) * 3];
    im0 += wv * c[0]; im1 += wv * c[1]; im2 += wv * c[2];
    wsum += wv;
  }
  out[r * 3 + 0] = im0 + (1.0f - wsum);
  out[r * 3 + 1] = im1 + (1.0f - wsum);
  out[r * 3 + 2] = im2 + (1.0f - wsum);
}

// ---------------------------------------------------------------------------
extern "C" void kernel_launch(void* const* d_in, const int* in_sizes, int n_in,
                              void* d_out, int out_size, void* d_ws, size_t ws_size,
                              hipStream_t stream) {
  const float* ro  = (const float*)d_in[0];
  const float* rd  = (const float*)d_in[1];
  const float* W1  = (const float*)d_in[2];
  const float* b1  = (const float*)d_in[3];
  const float* W2  = (const float*)d_in[4];
  const float* b2  = (const float*)d_in[5];
  const float* Wc1 = (const float*)d_in[6];
  const float* bc1 = (const float*)d_in[7];
  const float* Wc2 = (const float*)d_in[8];
  const float* bc2 = (const float*)d_in[9];
  const float* inv_s = (const float*)d_in[10];
  float* out = (float*)d_out;

  const int R = R_RAYS;
  float* ws     = (float*)d_ws;
  float* nearb  = ws;
  float* farb   = ws + R;
  float* sdistb = ws + 2 * R;
  float* zc     = ws + 3 * R;          // R*64
  float* sdfc   = zc + R * NSTEP;      // R*64
  float* deltas = sdfc + R * NSTEP;    // R*128
  float* zmid   = deltas + R * SFINE;  // R*128
  float* alpha  = zmid + R * SFINE;    // R*128
  float* colr   = alpha + R * SFINE;   // R*128*3

  prep_kernel<<<R / 256, 256, 0, stream>>>(ro, rd, zc, nearb, farb, sdistb);
  // 4096*64 points / 16 per tile / 8 tiles per block = 2048 blocks
  coarse_kernel<<<2048, 256, 0, stream>>>(ro, rd, W1, b1, W2, b2, zc, sdfc);
  pdf_kernel<<<R / 128, 128, 0, stream>>>(zc, sdfc, sdistb, deltas, zmid);
  // 4096*128 samples / 16 per tile / 4 tiles per block = 8192 blocks
  fine_kernel<<<8192, 128, 0, stream>>>(ro, rd, W1, b1, W2, b2, Wc1, bc1, Wc2, bc2,
                                        inv_s, zmid, deltas, alpha, colr);
  composite_kernel<<<R / 256, 256, 0, stream>>>(alpha, colr, out);
}
